// downLayer_52209622450786
// MI455X (gfx1250) — matrix operand order repaired and checked by reading, not compiled, not run on hardware
//
#include <hip/hip_runtime.h>
#include <hip/hip_bf16.h>

#define B_      2
#define L_      4096
#define DM_     192
#define DSSM_   384
#define CTOT_   646      // 384 + 262
#define CONVBC_ 262
#define HTOT_   24
#define HD_     64
#define NSTATE_ 512
#define CHUNK_  256
#define NC_     16

typedef __attribute__((ext_vector_type(16))) _Float16 v16h;
typedef __attribute__((ext_vector_type(8)))  _Float16 v8h;
typedef __attribute__((ext_vector_type(8)))  float    v8f;

// ---------------- CDNA5 WMMA helpers (16x16x32 f16 -> f32) ----------------
// A fragment from row-major tile [m][k] (ld multiple of 8 halfs):
// lane = m (mod 16); k = 16*(e/8) + 8*(lane/16) + (e&7) -> two aligned v8h reads
__device__ inline v16h ldsA(const _Float16* t, int r0, int ld) {
  const int lane = threadIdx.x & 31;
  const int m = r0 + (lane & 15);
  const int kb = (lane >> 4) << 3;
  const v8h lo = *(const v8h*)(t + m * ld + kb);
  const v8h hi = *(const v8h*)(t + m * ld + 16 + kb);
  return __builtin_shufflevector(lo, hi, 0, 1, 2, 3, 4, 5, 6, 7, 8, 9, 10, 11,
                                 12, 13, 14, 15);
}
// B fragment from N-major tile [n][k] (ld multiple of 8 halfs):
// lane = n (mod 16); k = 16*(e/8) + 8*(lane/16) + (e&7) -> two aligned v8h reads
__device__ inline v16h ldsBn(const _Float16* t, int c0, int ld) {
  const int lane = threadIdx.x & 31;
  const int n = c0 + (lane & 15);
  const int kb = (lane >> 4) << 3;
  const v8h lo = *(const v8h*)(t + n * ld + kb);
  const v8h hi = *(const v8h*)(t + n * ld + 16 + kb);
  return __builtin_shufflevector(lo, hi, 0, 1, 2, 3, 4, 5, 6, 7, 8, 9, 10, 11,
                                 12, 13, 14, 15);
}
__device__ inline v8f wmma32(v16h a, v16h b, v8f c) {
  return __builtin_amdgcn_wmma_f32_16x16x32_f16(false, a, false, b, (short)0, c,
                                                false, false);
}
// C/D layout: VGPR r -> row r + 8*(lane/16); col = lane & 15
__device__ inline void storeC(float* base, long ld, int m0, int n0, v8f acc) {
  const int lane = threadIdx.x & 31;
  const int col = n0 + (lane & 15);
  const int rb = m0 + ((lane >> 4) << 3);
#pragma unroll
  for (int r = 0; r < 8; ++r) base[(long)(rb + r) * ld + col] = acc[r];
}
__device__ inline v8f loadC(const float* base, long ld, int m0, int n0) {
  const int lane = threadIdx.x & 31;
  const int col = n0 + (lane & 15);
  const int rb = m0 + ((lane >> 4) << 3);
  v8f a;
#pragma unroll
  for (int r = 0; r < 8; ++r) a[r] = base[(long)(rb + r) * ld + col];
  return a;
}

// ---------------- generic GEMM: C[M,N] = A[M,Kd](f32) @ W[Kd,N](f32) -------
// 128 threads = 4 waves; 64x64 tile; wave w does rows [16w,16w+16)
__global__ __launch_bounds__(128) void k_gemm(const float* __restrict__ A,
                                              const float* __restrict__ W,
                                              float* __restrict__ C, int Kd,
                                              int N, int ldC, int c0) {
  __shared__ alignas(16) _Float16 At[64][40];
  __shared__ alignas(16) _Float16 Bt[64][40];  // [n][k]
  const int tid = threadIdx.x, w = tid >> 5;
  const int m0 = blockIdx.x * 64, n0 = blockIdx.y * 64;
  v8f acc[4] = {};
  for (int k0 = 0; k0 < Kd; k0 += 32) {
    // A tile: 64x32 f32 -> f16, per-thread 8 contiguous elements
    for (int q = 0; q < 2; ++q) {
      int g = tid + 128 * q;            // 0..255
      int r = g >> 2, k8 = (g & 3) * 8;
      const float* src = A + (long)(m0 + r) * Kd + k0 + k8;
      v8h h;
#pragma unroll
      for (int e = 0; e < 8; ++e) h[e] = (_Float16)src[e];
      *(v8h*)(&At[r][k8]) = h;
    }
    // W tile -> Bt[n][k]; read coalesced along n, scatter into LDS
    for (int q = 0; q < 16; ++q) {
      int idx = tid + 128 * q;
      int k = idx >> 6, n = idx & 63;
      float v = (n0 + n < N) ? W[(long)(k0 + k) * N + n0 + n] : 0.f;
      Bt[n][k] = (_Float16)v;
    }
    __syncthreads();
    v16h a = ldsA(&At[0][0], 16 * w, 40);
#pragma unroll
    for (int t = 0; t < 4; ++t) {
      v16h bb = ldsBn(&Bt[0][0], 16 * t, 40);
      acc[t] = wmma32(a, bb, acc[t]);
    }
    __syncthreads();
  }
  const int lane = tid & 31;
  const int col = lane & 15;
  const int rb = m0 + 16 * w + ((lane >> 4) << 3);
  for (int t = 0; t < 4; ++t) {
    int cc = n0 + 16 * t + col;
    if (cc < N)
#pragma unroll
      for (int r = 0; r < 8; ++r) C[(long)(rb + r) * ldC + c0 + cc] = acc[t][r];
  }
}

// ------------- depthwise 3x3 conv + SiLU: pre(B,L,646) -> xbc(B,646,L) -----
__global__ __launch_bounds__(256) void k_conv(const float* __restrict__ pre,
                                              const float* __restrict__ wxs,
                                              const float* __restrict__ bxs,
                                              const float* __restrict__ wbc,
                                              const float* __restrict__ bbc,
                                              float* __restrict__ xbc) {
  const int l = blockIdx.x * 256 + threadIdx.x;
  const int ch = blockIdx.y, b = blockIdx.z;
  const int y = l >> 6, x = l & 63;
  const float* wgt;
  float s;
  if (ch < DSSM_) { wgt = wxs + ch * 9; s = bxs[ch]; }
  else            { wgt = wbc + (ch - DSSM_) * 9; s = bbc[ch - DSSM_]; }
#pragma unroll
  for (int ky = 0; ky < 3; ++ky) {
    int yy = y + ky - 1;
    if (yy < 0 || yy > 63) continue;
#pragma unroll
    for (int kx = 0; kx < 3; ++kx) {
      int xx = x + kx - 1;
      if (xx < 0 || xx > 63) continue;
      s += pre[((long)(b * L_ + yy * 64 + xx)) * CTOT_ + ch] * wgt[ky * 3 + kx];
    }
  }
  xbc[((long)b * CTOT_ + ch) * L_ + l] = s / (1.f + expf(-s));
}

// ------- build 4-direction tensors: X(f16), Bm(f16), Cm(f16), dtraw(f32) ---
__global__ __launch_bounds__(256) void k_build(const float* __restrict__ xbc,
                                               _Float16* __restrict__ X,
                                               _Float16* __restrict__ Bmh,
                                               _Float16* __restrict__ Cmh,
                                               float* __restrict__ dtraw) {
  const int l = blockIdx.x * 256 + threadIdx.x;
  const int ch = blockIdx.y;
  const int b = blockIdx.z >> 2, d = blockIdx.z & 3;
  int src;
  if (d == 0) src = l;
  else if (d == 1) { int w0 = l >> 6, h0 = l & 63; src = h0 * 64 + w0; }
  else if (d == 2) src = (L_ - 1) - l;
  else { int lf = (L_ - 1) - l; int w0 = lf >> 6, h0 = lf & 63; src = h0 * 64 + w0; }
  const float v = xbc[((long)b * CTOT_ + ch) * L_ + src];
  const long pos = (long)b * L_ + l;
  if (ch < DSSM_) {
    int ht = d * 6 + (ch >> 6), p = ch & 63;
    X[(pos * HTOT_ + ht) * HD_ + p] = (_Float16)v;
  } else if (ch < DSSM_ + 128) {
    Bmh[pos * NSTATE_ + d * 128 + (ch - DSSM_)] = (_Float16)v;
  } else if (ch < DSSM_ + 256) {
    Cmh[pos * NSTATE_ + d * 128 + (ch - DSSM_ - 128)] = (_Float16)v;
  } else {
    dtraw[pos * HTOT_ + d * 6 + (ch - DSSM_ - 256)] = v;
  }
}

// ------- softplus(dt)+bias, dA cumsum per chunk, chunk decay ---------------
__global__ __launch_bounds__(256) void k_dt(const float* __restrict__ dtraw,
                                            const float* __restrict__ dtb,
                                            const float* __restrict__ Alog,
                                            float* __restrict__ dtc,
                                            float* __restrict__ acum,
                                            float* __restrict__ cdecay) {
  const int c = blockIdx.x, ht = blockIdx.y, b = blockIdx.z;
  const int i = threadIdx.x;
  __shared__ float s[256];
  const long gi = ((long)(b * L_ + c * CHUNK_ + i)) * HTOT_ + ht;
  float x = dtraw[gi] + dtb[ht];
  float dt = (x > 20.f) ? x : log1pf(expf(x));
  dtc[gi] = dt;
  s[i] = dt * (-expf(Alog[ht]));
  __syncthreads();
  for (int off = 1; off < 256; off <<= 1) {
    float add = (i >= off) ? s[i - off] : 0.f;
    __syncthreads();
    s[i] += add;
    __syncthreads();
  }
  acum[gi] = s[i];
  if (i == 255) cdecay[(b * NC_ + c) * HTOT_ + ht] = expf(s[i]);
}

// ------- CB[i,j] = sum_n C[i,n]*B[j,n], per (b,chunk), 64x64 tiles ---------
__global__ __launch_bounds__(128) void k_cb(const _Float16* __restrict__ Cmh,
                                            const _Float16* __restrict__ Bmh,
                                            float* __restrict__ CB) {
  const int bc = blockIdx.y, b = bc >> 4, c = bc & 15;
  const int i0 = (blockIdx.x >> 2) * 64, j0 = (blockIdx.x & 3) * 64;
  __shared__ alignas(16) _Float16 At[64][40];
  __shared__ alignas(16) _Float16 Bt[64][40];  // [j][n(k)] -- direct row copy
  const int tid = threadIdx.x, w = tid >> 5;
  const long row0 = (long)b * L_ + c * CHUNK_;
  v8f acc[4] = {};
  for (int n0 = 0; n0 < NSTATE_; n0 += 32) {
    // both tiles are contiguous 8-half runs in global: 128-bit loads + stores
    for (int q = 0; q < 2; ++q) {
      int g = tid + 128 * q;            // 0..255 -> (row, k8)
      int r = g >> 2, k8 = (g & 3) * 8;
      *(v8h*)(&At[r][k8]) =
          *(const v8h*)(Cmh + (row0 + i0 + r) * NSTATE_ + n0 + k8);
      *(v8h*)(&Bt[r][k8]) =
          *(const v8h*)(Bmh + (row0 + j0 + r) * NSTATE_ + n0 + k8);
    }
    __syncthreads();
    v16h a = ldsA(&At[0][0], 16 * w, 40);
#pragma unroll
    for (int t = 0; t < 4; ++t) {
      v16h bb = ldsBn(&Bt[0][0], 16 * t, 40);  // B^T handled by N-major layout
      acc[t] = wmma32(a, bb, acc[t]);
    }
    __syncthreads();
  }
  float* base = CB + ((long)bc << 16) + i0 * 256 + j0;
  for (int t = 0; t < 4; ++t) storeC(base, 256, 16 * w, 16 * t, acc[t]);
}

// ------- intra-chunk y[i,p] = sum_j M[i,j]*x[j,p]  (M built on the fly) ----
__global__ __launch_bounds__(256) void k_ssd_y(const _Float16* __restrict__ X,
                                               const float* __restrict__ CB,
                                               const float* __restrict__ acum,
                                               const float* __restrict__ dtc,
                                               float* __restrict__ Y) {
  const int ht = blockIdx.x, bc = blockIdx.y, b = bc >> 4, c = bc & 15;
  __shared__ float sA[256], sDt[256];
  __shared__ alignas(16) _Float16 Am[256][40];
  __shared__ alignas(16) _Float16 Bt[64][40];  // [p][j(k)]
  const int tid = threadIdx.x, w = tid >> 5;
  const long row0 = (long)b * L_ + c * CHUNK_;
  sA[tid] = acum[(row0 + tid) * HTOT_ + ht];
  sDt[tid] = dtc[(row0 + tid) * HTOT_ + ht];
  __syncthreads();
  v8f acc[2][4] = {};
  for (int jb = 0; jb < 8; ++jb) {
    const int j0 = jb * 32;
    const int i = tid;  // each thread builds one row of the M tile
    const float ea = sA[i];
    const float* cbrow = CB + ((long)bc << 16) + (i << 8) + j0;
#pragma unroll
    for (int k8 = 0; k8 < 32; k8 += 8) {
      v8h h;
#pragma unroll
      for (int e = 0; e < 8; ++e) {
        int j = j0 + k8 + e;
        float v = (i >= j) ? cbrow[k8 + e] * expf(ea - sA[j]) * sDt[j] : 0.f;
        h[e] = (_Float16)v;
      }
      *(v8h*)(&Am[i][k8]) = h;
    }
    // x tile: read 8 contiguous p per thread, scatter transposed into Bt[p][k]
    {
      int g = tid;                      // 0..255 -> (k, p8)
      int k = g >> 3, p8 = (g & 7) * 8;
      v8h xv = *(const v8h*)(X + ((row0 + j0 + k) * HTOT_ + ht) * HD_ + p8);
#pragma unroll
      for (int e = 0; e < 8; ++e) Bt[p8 + e][k] = xv[e];
    }
    __syncthreads();
#pragma unroll
    for (int s = 0; s < 2; ++s) {
      v16h a = ldsA(&Am[0][0], w * 32 + s * 16, 40);
#pragma unroll
      for (int t = 0; t < 4; ++t) {
        v16h bb = ldsBn(&Bt[0][0], t * 16, 40);
        acc[s][t] = wmma32(a, bb, acc[s][t]);
      }
    }
    __syncthreads();
  }
  float* outb = Y + (row0 * HTOT_ + ht) * HD_;
  for (int s = 0; s < 2; ++s)
    for (int t = 0; t < 4; ++t)
      storeC(outb, (long)HTOT_ * HD_, w * 32 + s * 16, t * 16, acc[s][t]);
}

// ------- states[p,n] = sum_j (decay_end*dt*x)[j,p] * B[j,n] ----------------
__global__ __launch_bounds__(128) void k_state(const _Float16* __restrict__ X,
                                               const _Float16* __restrict__ Bmh,
                                               const float* __restrict__ acum,
                                               const float* __restrict__ dtc,
                                               float* __restrict__ S) {
  const int nb = blockIdx.x, ht = blockIdx.y, bc = blockIdx.z;
  const int b = bc >> 4, c = bc & 15;
  __shared__ float sAc[256], sW[256];
  __shared__ alignas(16) _Float16 At[64][40];   // [p][j(k)]
  __shared__ alignas(16) _Float16 Bt[128][40];  // [n][j(k)]
  const int tid = threadIdx.x, w = tid >> 5;
  const long row0 = (long)b * L_ + c * CHUNK_;
  for (int q = 0; q < 2; ++q) {
    int i = tid + 128 * q;
    sAc[i] = acum[(row0 + i) * HTOT_ + ht];
  }
  __syncthreads();
  const float alast = sAc[255];
  for (int q = 0; q < 2; ++q) {
    int i = tid + 128 * q;
    sW[i] = expf(alast - sAc[i]) * dtc[(row0 + i) * HTOT_ + ht];
  }
  __syncthreads();
  v8f acc[8] = {};
  for (int jb = 0; jb < 8; ++jb) {
    const int j0 = jb * 32;
    // A tile: read x rows (contig p), scale, scatter transposed
    for (int q = 0; q < 2; ++q) {
      int g = tid + 128 * q;            // 0..255 -> (k, p8)
      int k = g >> 3, p8 = (g & 7) * 8;
      v8h xv = *(const v8h*)(X + ((row0 + j0 + k) * HTOT_ + ht) * HD_ + p8);
      float wj = sW[j0 + k];
#pragma unroll
      for (int e = 0; e < 8; ++e) At[p8 + e][k] = (_Float16)((float)xv[e] * wj);
    }
    // B tile: read B rows (contig n), scatter transposed into Bt[n][k]
    for (int q = 0; q < 4; ++q) {
      int g = tid + 128 * q;            // 0..511 -> (k, n8)
      int k = g >> 4, n8 = (g & 15) * 8;
      v8h bv = *(const v8h*)(Bmh + (row0 + j0 + k) * NSTATE_ + nb * 128 + n8);
#pragma unroll
      for (int e = 0; e < 8; ++e) Bt[n8 + e][k] = bv[e];
    }
    __syncthreads();
    v16h a = ldsA(&At[0][0], 16 * w, 40);
#pragma unroll
    for (int t = 0; t < 8; ++t) {
      v16h bb = ldsBn(&Bt[0][0], 16 * t, 40);
      acc[t] = wmma32(a, bb, acc[t]);
    }
    __syncthreads();
  }
  float* base = S + (((long)bc * HTOT_ + ht) * HD_) * NSTATE_ + nb * 128;
  for (int t = 0; t < 8; ++t) storeC(base, NSTATE_, 16 * w, 16 * t, acc[t]);
}

// ------- inter-chunk recurrence: states -> prev (in place), 16 steps -------
__global__ __launch_bounds__(256) void k_scan(float* __restrict__ S,
                                              const float* __restrict__ cdecay) {
  long t = (long)blockIdx.x * 256 + threadIdx.x;  // over B*HTOT*HD*NSTATE
  const int n = (int)(t & 511);
  long r = t >> 9;
  const int p = (int)(r & 63);
  r >>= 6;
  const int ht = (int)(r % HTOT_);
  const int b = (int)(r / HTOT_);
  float Sv = 0.f;
  for (int c = 0; c < NC_; ++c) {
    long idx = (((long)(b * NC_ + c) * HTOT_ + ht) * HD_ + p) * NSTATE_ + n;
    float st = S[idx];
    S[idx] = Sv;  // prev state entering chunk c
    Sv = Sv * cdecay[(b * NC_ + c) * HTOT_ + ht] + st;
  }
}

// ------- y += (exp(acum)*C)[i,n] @ prev[n,p] -------------------------------
__global__ __launch_bounds__(256) void k_yprev(const _Float16* __restrict__ Cmh,
                                               const float* __restrict__ Sprev,
                                               const float* __restrict__ acum,
                                               float* __restrict__ Y) {
  const int ht = blockIdx.x, bc = blockIdx.y, b = bc >> 4, c = bc & 15;
  __shared__ float sEa[256];
  __shared__ alignas(16) _Float16 Am[256][40];
  __shared__ alignas(16) _Float16 Bt[64][40];  // [p][n(k)] -- direct row copy
  const int tid = threadIdx.x, w = tid >> 5;
  const long row0 = (long)b * L_ + c * CHUNK_;
  sEa[tid] = expf(acum[(row0 + tid) * HTOT_ + ht]);
  __syncthreads();
  float* yb = Y + (row0 * HTOT_ + ht) * HD_;
  v8f acc[2][4];
  for (int s = 0; s < 2; ++s)
    for (int t = 0; t < 4; ++t)
      acc[s][t] = loadC(yb, (long)HTOT_ * HD_, w * 32 + s * 16, t * 16);
  const float* pv = Sprev + (((long)bc * HTOT_ + ht) * HD_) * NSTATE_;
  for (int n0 = 0; n0 < NSTATE_; n0 += 32) {
    const float ea = sEa[tid];
    const _Float16* crow = Cmh + (row0 + tid) * NSTATE_ + n0;
#pragma unroll
    for (int k8 = 0; k8 < 32; k8 += 8) {
      v8h cv = *(const v8h*)(crow + k8);
      v8h h;
#pragma unroll
      for (int e = 0; e < 8; ++e) h[e] = (_Float16)((float)cv[e] * ea);
      *(v8h*)(&Am[tid][k8]) = h;
    }
    {
      int g = tid;                      // 0..255 -> (p, k8): contiguous copy
      int p = g >> 2, k8 = (g & 3) * 8;
      const float* src = pv + (long)p * NSTATE_ + n0 + k8;
      v8h h;
#pragma unroll
      for (int e = 0; e < 8; ++e) h[e] = (_Float16)src[e];
      *(v8h*)(&Bt[p][k8]) = h;
    }
    __syncthreads();
#pragma unroll
    for (int s = 0; s < 2; ++s) {
      v16h a = ldsA(&Am[0][0], w * 32 + s * 16, 40);
#pragma unroll
      for (int t = 0; t < 4; ++t) {
        v16h bb = ldsBn(&Bt[0][0], t * 16, 40);
        acc[s][t] = wmma32(a, bb, acc[s][t]);
      }
    }
    __syncthreads();
  }
  for (int s = 0; s < 2; ++s)
    for (int t = 0; t < 4; ++t)
      storeC(yb, (long)HTOT_ * HD_, w * 32 + s * 16, t * 16, acc[s][t]);
}

// ------- combine 4 dirs + D*x + SiLU(z) gate + RMS norm --------------------
__global__ __launch_bounds__(128) void k_combine(const float* __restrict__ Y,
                                                 const _Float16* __restrict__ X,
                                                 const float* __restrict__ z,
                                                 const float* __restrict__ Ds,
                                                 const float* __restrict__ nw,
                                                 float* __restrict__ gn) {
  const int bl = blockIdx.x;
  const int b = bl >> 12, l = bl & 4095;
  const int h0 = l >> 6, w0 = l & 63;
  const int tid = threadIdx.x;
  __shared__ float gbuf[DSSM_];
  __shared__ float red[128];
  int src[4];
  src[0] = l;
  src[1] = w0 * 64 + h0;
  src[2] = 4095 - l;
  src[3] = 4095 - (w0 * 64 + h0);
  float ssq = 0.f;
  for (int q = 0; q < 3; ++q) {
    int ch = tid + 128 * q;
    int hh = ch >> 6, p = ch & 63;
    float acc = 0.f;
#pragma unroll
    for (int d = 0; d < 4; ++d) {
      int ht = d * 6 + hh;
      long idx = (((long)b * L_ + src[d]) * HTOT_ + ht) * HD_ + p;
      acc += Y[idx] + Ds[ht] * (float)X[idx];
    }
    float zv = z[(long)bl * DSSM_ + ch];
    float g = acc * (zv / (1.f + expf(-zv)));
    gbuf[ch] = g;
    ssq += g * g;
  }
  red[tid] = ssq;
  __syncthreads();
  for (int s = 64; s > 0; s >>= 1) {
    if (tid < s) red[tid] += red[tid + s];
    __syncthreads();
  }
  const float r = rsqrtf(red[0] / (float)DSSM_ + 1e-5f);
  for (int q = 0; q < 3; ++q) {
    int ch = tid + 128 * q;
    gn[(long)bl * DSSM_ + ch] = gbuf[ch] * r * nw[ch];
  }
}

// ---------------------------------------------------------------------------
extern "C" void kernel_launch(void* const* d_in, const int* in_sizes, int n_in,
                              void* d_out, int out_size, void* d_ws,
                              size_t ws_size, hipStream_t stream) {
  (void)in_sizes; (void)n_in; (void)out_size; (void)ws_size;
  const float* u[2] = {(const float*)d_in[0], (const float*)d_in[1]};
  const float* ux[2] = {(const float*)d_in[2], (const float*)d_in[3]};
  const float* Wskip = (const float*)d_in[4];
  const float* Wxs = (const float*)d_in[5];
  const float* Wbcdt = (const float*)d_in[6];
  const float* cxw = (const float*)d_in[7];
  const float* cxb = (const float*)d_in[8];
  const float* cbw = (const float*)d_in[9];
  const float* cbb = (const float*)d_in[10];
  const float* dtb = (const float*)d_in[11];
  const float* Alog = (const float*)d_in[12];
  const float* Ds = (const float*)d_in[13];
  const float* nw = (const float*)d_in[14];
  const float* Wout = (const float*)d_in[15];

  char* ws = (char*)d_ws;
  size_t off = 0;
  auto take = [&](size_t bytes) -> void* {
    size_t o = (off + 255) & ~(size_t)255;
    off = o + bytes;
    return (void*)(ws + o);
  };
  const size_t ML = (size_t)B_ * L_;  // 8192 rows
  float*    z     = (float*)take(ML * DSSM_ * 4);
  float*    pre   = (float*)take(ML * CTOT_ * 4);
  float*    xbc   = (float*)take(ML * CTOT_ * 4);
  _Float16* X     = (_Float16*)take(ML * HTOT_ * HD_ * 2);
  _Float16* Bmh   = (_Float16*)take(ML * NSTATE_ * 2);
  _Float16* Cmh   = (_Float16*)take(ML * NSTATE_ * 2);
  float*    dtraw = (float*)take(ML * HTOT_ * 4);
  float*    dtc   = (float*)take(ML * HTOT_ * 4);
  float*    acum  = (float*)take(ML * HTOT_ * 4);
  float*    cdec  = (float*)take((size_t)B_ * NC_ * HTOT_ * 4);
  float*    CB    = (float*)take((size_t)B_ * NC_ * CHUNK_ * CHUNK_ * 4);
  float*    Y     = (float*)take(ML * HTOT_ * HD_ * 4);
  float*    S     = (float*)take((size_t)B_ * NC_ * HTOT_ * HD_ * NSTATE_ * 4);
  float*    gn    = (float*)take(ML * DSSM_ * 4);

  for (int br = 0; br < 2; ++br) {
    float* outb = (float*)d_out + (size_t)br * ML * DM_;
    // input projections
    k_gemm<<<dim3(128, 6), 128, 0, stream>>>(u[br], Wskip, z, DM_, DSSM_, DSSM_, 0);
    k_gemm<<<dim3(128, 6), 128, 0, stream>>>(u[br], Wxs, pre, DM_, DSSM_, CTOT_, 0);
    k_gemm<<<dim3(128, 5), 128, 0, stream>>>(ux[br], Wbcdt, pre, DM_, CONVBC_, CTOT_, DSSM_);
    // depthwise conv + silu, then 4-direction rearrange
    k_conv<<<dim3(16, CTOT_, B_), 256, 0, stream>>>(pre, cxw, cxb, cbw, cbb, xbc);
    k_build<<<dim3(16, CTOT_, B_ * 4), 256, 0, stream>>>(xbc, X, Bmh, Cmh, dtraw);
    // dt softplus + per-chunk cumsum
    k_dt<<<dim3(NC_, HTOT_, B_), 256, 0, stream>>>(dtraw, dtb, Alog, dtc, acum, cdec);
    // SSD
    k_cb<<<dim3(16, B_ * NC_), 128, 0, stream>>>(Cmh, Bmh, CB);
    k_ssd_y<<<dim3(HTOT_, B_ * NC_), 256, 0, stream>>>(X, CB, acum, dtc, Y);
    k_state<<<dim3(4, HTOT_, B_ * NC_), 128, 0, stream>>>(X, Bmh, acum, dtc, S);
    k_scan<<<(B_ * HTOT_ * HD_ * NSTATE_) / 256, 256, 0, stream>>>(S, cdec);
    k_yprev<<<dim3(HTOT_, B_ * NC_), 256, 0, stream>>>(Cmh, S, acum, Y);
    // merge directions, gate, norm, output projection
    k_combine<<<ML, 128, 0, stream>>>(Y, X, z, Ds, nw, gn);
    k_gemm<<<dim3(128, 3), 128, 0, stream>>>(gn, Wout, outb, DSSM_, DM_, DM_, 0);
  }
}
